// MultiHeadedAttention_64862596104203
// MI455X (gfx1250) — compile-verified, matrix operand order checked by reading, not run on hardware
//
#include <hip/hip_runtime.h>
#include <stdint.h>

typedef __bf16 bf16;
typedef __bf16 v16bf __attribute__((ext_vector_type(16)));
typedef __bf16 v8bf  __attribute__((ext_vector_type(8)));
typedef float  v8f   __attribute__((ext_vector_type(8)));

#define DIM   1024
#define SEQ   2048
#define NH    16
#define HDIM  64
#define MB    8192          // B*S rows

// ---------------- CDNA5 helpers ----------------

__device__ __forceinline__ uint32_t lds_off_u32(const void* p) {
  // generic pointer to LDS: low 32 bits are the LDS byte address (ISA 10.2 aperture rules)
  return (uint32_t)(uintptr_t)p;
}

// async global -> LDS copy, 16B per lane (ASYNCcnt-tracked, gfx1250)
__device__ __forceinline__ void async_copy_b128(void* lds_dst, const void* gsrc) {
  asm volatile("global_load_async_to_lds_b128 %0, %1, off"
               :: "v"(lds_off_u32(lds_dst)), "v"(gsrc)
               : "memory");
}
__device__ __forceinline__ void wait_async0() {
  asm volatile("s_wait_asynccnt 0x0" ::: "memory");
}
__device__ __forceinline__ void wait_ds0() {
  asm volatile("s_wait_dscnt 0x0" ::: "memory");
}

union FragBF { v16bf v; v8bf h[2]; };

// build a 16-element bf16 fragment from two contiguous 16B LDS chunks
__device__ __forceinline__ v16bf load_frag(const bf16* c0, const bf16* c1) {
  FragBF f;
  f.h[0] = *(const v8bf*)c0;
  f.h[1] = *(const v8bf*)c1;
  return f.v;
}

__device__ __forceinline__ v8f wmma_bf16(v16bf a, v16bf b, v8f c) {
  return __builtin_amdgcn_wmma_f32_16x16x32_bf16(false, a, false, b, (short)0, c,
                                                 false, false);
}

// ---------------- conversion kernels ----------------

__global__ void cvt_bf16_kernel(const float* __restrict__ src, bf16* __restrict__ dst, int n) {
  int i = blockIdx.x * blockDim.x + threadIdx.x;
  int stride = gridDim.x * blockDim.x;
  for (; i < n; i += stride) dst[i] = (bf16)src[i];
}

// w[k*N + n] (f32, KxN) -> wt[n*K + k] (bf16, NxK)
__global__ void cvt_transpose_kernel(const float* __restrict__ w, bf16* __restrict__ wt,
                                     int K, int N) {
  int i = blockIdx.x * blockDim.x + threadIdx.x;
  int total = K * N;
  int stride = gridDim.x * blockDim.x;
  for (; i < total; i += stride) {
    int k = i / N, n = i - k * N;
    wt[(size_t)n * K + k] = (bf16)w[i];
  }
}

// ---------------- tiled WMMA GEMM: C = A(MxK) * Bt(NxK)^T + bias ----------------
// Block tile 128x256, 8 waves as 2(M) x 4(N), 64x64 per wave (16 WMMA accumulators).
// mode 0: write bf16 head-split Q  [b,h,s,d]
// mode 1: n<DIM -> K [b,h,l,d];  n>=DIM -> V^T [b,h,d,l]
// mode 2: write f32 row-major to outf (final projection)

#define BM 128
#define BN 256
#define BK 32

__global__ __launch_bounds__(256)
void gemm_bf16_wmma_kernel(const bf16* __restrict__ A, const bf16* __restrict__ Bt,
                           const float* __restrict__ bias,
                           bf16* __restrict__ outb0, bf16* __restrict__ outb1,
                           float* __restrict__ outf,
                           int M, int N, int K, int mode) {
  __shared__ bf16 Asm[2][BM * BK];   // 8 KB / buffer
  __shared__ bf16 Bsm[2][BN * BK];   // 16 KB / buffer

  const int tid  = threadIdx.x;
  const int wave = tid >> 5;
  const int lane = tid & 31;
  const int hi   = lane >> 4;
  const int ln   = lane & 15;
  const int wm   = wave >> 2;     // 0..1 : M quadrant (64 rows)
  const int wn   = wave & 3;      // 0..3 : N quadrant (64 cols)
  const int m0   = blockIdx.y * BM;
  const int n0   = blockIdx.x * BN;

  v8f acc[4][4];
#pragma unroll
  for (int i = 0; i < 4; ++i)
#pragma unroll
    for (int j = 0; j < 4; ++j) acc[i][j] = {};

  auto stage = [&](int buf, int k0) {
    // A tile: 128 rows x 64B = 512 x b128 ; B tile: 256 rows x 64B = 1024 x b128
#pragma unroll
    for (int rep = 0; rep < 2; ++rep) {
      int c   = tid + rep * 256;
      int row = c >> 2;              // 4 x 16B chunks per 64B row
      int co  = (c & 3) * 16;        // byte offset within row
      async_copy_b128(&Asm[buf][row * BK + co / 2],
                      (const char*)A + ((size_t)(m0 + row) * K + k0) * 2 + co);
    }
#pragma unroll
    for (int rep = 0; rep < 4; ++rep) {
      int c   = tid + rep * 256;
      int row = c >> 2;
      int co  = (c & 3) * 16;
      async_copy_b128(&Bsm[buf][row * BK + co / 2],
                      (const char*)Bt + ((size_t)(n0 + row) * K + k0) * 2 + co);
    }
  };

  int buf = 0;
  stage(0, 0);
  const int KT = K / BK;
  for (int kt = 0; kt < KT; ++kt) {
    wait_async0();
    __syncthreads();
    if (kt + 1 < KT) stage(buf ^ 1, (kt + 1) * BK);   // overlap copy with WMMA

    v16bf afr[4], bfr[4];
#pragma unroll
    for (int mt = 0; mt < 4; ++mt) {
      int r = wm * 64 + mt * 16 + ln;
      afr[mt] = load_frag(&Asm[buf][r * BK + hi * 8],
                          &Asm[buf][r * BK + 16 + hi * 8]);
    }
#pragma unroll
    for (int nt = 0; nt < 4; ++nt) {
      int r = wn * 64 + nt * 16 + ln;
      bfr[nt] = load_frag(&Bsm[buf][r * BK + hi * 8],
                          &Bsm[buf][r * BK + 16 + hi * 8]);
    }
#pragma unroll
    for (int mt = 0; mt < 4; ++mt)
#pragma unroll
      for (int nt = 0; nt < 4; ++nt)
        acc[mt][nt] = wmma_bf16(afr[mt], bfr[nt], acc[mt][nt]);

    __syncthreads();
    buf ^= 1;
  }

  // epilogue: C layout -> VGPR r holds row (r + 8*hi), col = ln within 16x16 tile
#pragma unroll
  for (int mt = 0; mt < 4; ++mt) {
#pragma unroll
    for (int nt = 0; nt < 4; ++nt) {
#pragma unroll
      for (int r = 0; r < 8; ++r) {
        int gm = m0 + wm * 64 + mt * 16 + r + 8 * hi;
        int gn = n0 + wn * 64 + nt * 16 + ln;
        float v = acc[mt][nt][r] + bias[gn];
        if (mode == 2) {
          outf[(size_t)gm * N + gn] = v;
        } else if (mode == 0) {
          int bb = gm >> 11, ss = gm & 2047;
          int hh = gn >> 6,  dd = gn & 63;
          outb0[((size_t)(bb * NH + hh) * SEQ + ss) * HDIM + dd] = (bf16)v;
        } else {
          int bb = gm >> 11, ll = gm & 2047;
          if (gn < DIM) {
            int hh = gn >> 6, dd = gn & 63;
            outb0[((size_t)(bb * NH + hh) * SEQ + ll) * HDIM + dd] = (bf16)v;
          } else {
            int n2 = gn - DIM;
            int hh = n2 >> 6, dd = n2 & 63;
            outb1[((size_t)(bb * NH + hh) * HDIM + dd) * SEQ + ll] = (bf16)v;
          }
        }
      }
    }
  }
}

// ---------------- flash attention (online softmax), 64-wide L chunks ----------------
// Q [b,h,s,64] bf16 ; K [b,h,l,64] bf16 (B-frag ready for Q.K^T) ;
// Vt [b,h,64,l] bf16 (B-frag ready for P.V) ; out bf16 [b,s,1024]

#define LCH 64

__global__ __launch_bounds__(256)
void attn_wmma_kernel(const bf16* __restrict__ Q, const bf16* __restrict__ Kh,
                      const bf16* __restrict__ Vt, bf16* __restrict__ Ob) {
  __shared__ bf16 Ksm[2][LCH * 64];   // 64 l-rows x 64 d      (8 KB / buffer)
  __shared__ bf16 Vsm[2][64 * LCH];   // 64 d-rows x 64 l      (8 KB / buffer)
  __shared__ bf16 Psm[8][16 * LCH];   // per-wave P bounce     (16 KB total)

  const int tid  = threadIdx.x;
  const int wave = tid >> 5;
  const int lane = tid & 31;
  const int hi   = lane >> 4;
  const int ln   = lane & 15;
  const int s0   = blockIdx.x * 128;
  const int h    = blockIdx.y;
  const int b    = blockIdx.z;
  const size_t hb = (size_t)(b * NH + h);

  // Q fragments for this wave's 16 rows (K-dim d=64 -> 2 fragments), straight from global
  const bf16* qbase = Q + (hb * SEQ + (s0 + wave * 16 + ln)) * HDIM;
  v16bf qfr[2];
#pragma unroll
  for (int f = 0; f < 2; ++f) {
    FragBF fr;
    fr.h[0] = *(const v8bf*)(qbase + f * 32 + hi * 8);
    fr.h[1] = *(const v8bf*)(qbase + f * 32 + 16 + hi * 8);
    qfr[f] = fr.v;
  }

  v8f O[4];
#pragma unroll
  for (int dt = 0; dt < 4; ++dt) O[dt] = {};
  float rmax[8], rsum[8];
#pragma unroll
  for (int r = 0; r < 8; ++r) { rmax[r] = -3.0e38f; rsum[r] = 0.0f; }

  const char* kbase = (const char*)(Kh + hb * SEQ * HDIM);
  const char* vbase = (const char*)(Vt + hb * HDIM * SEQ);

  auto stage = [&](int buf, int l0) {
    // K tile: 64 rows x 128B = 512 chunks ; V tile: 64 rows x 128B = 512 chunks
#pragma unroll
    for (int rep = 0; rep < 2; ++rep) {
      int c = tid + rep * 256;
      int row = c >> 3, co = (c & 7) * 16;
      async_copy_b128(&Ksm[buf][row * 64 + co / 2],
                      kbase + (size_t)(l0 + row) * 128 + co);
    }
#pragma unroll
    for (int rep = 0; rep < 2; ++rep) {
      int c = tid + rep * 256;
      int row = c >> 3, co = (c & 7) * 16;
      async_copy_b128(&Vsm[buf][row * LCH + co / 2],
                      vbase + (size_t)row * (SEQ * 2) + (size_t)l0 * 2 + co);
    }
  };

  int buf = 0;
  stage(0, 0);
  const int LT = SEQ / LCH;
  for (int lt = 0; lt < LT; ++lt) {
    wait_async0();
    __syncthreads();
    if (lt + 1 < LT) stage(buf ^ 1, (lt + 1) * LCH);

    // scores S = Q . K^T  (four 16x16 l-tiles per 64-chunk)
    v8f S[4];
#pragma unroll
    for (int nt = 0; nt < 4; ++nt) {
      v8f sa = {};
#pragma unroll
      for (int kk = 0; kk < 2; ++kk) {
        int r = nt * 16 + ln;
        v16bf bfr = load_frag(&Ksm[buf][r * 64 + kk * 32 + hi * 8],
                              &Ksm[buf][r * 64 + kk * 32 + 16 + hi * 8]);
        sa = wmma_bf16(qfr[kk], bfr, sa);
      }
      S[nt] = sa;
    }

    // online softmax: rows r+8*hi live across the 16 lanes of this half-wave
    float nm[8], corr[8];
#pragma unroll
    for (int r = 0; r < 8; ++r) {
      float v = fmaxf(fmaxf(S[0][r], S[1][r]), fmaxf(S[2][r], S[3][r])) * 0.125f;
      v = fmaxf(v, __shfl_xor(v, 1, 32));
      v = fmaxf(v, __shfl_xor(v, 2, 32));
      v = fmaxf(v, __shfl_xor(v, 4, 32));
      v = fmaxf(v, __shfl_xor(v, 8, 32));
      nm[r]   = fmaxf(rmax[r], v);
      corr[r] = __expf(rmax[r] - nm[r]);
      rmax[r] = nm[r];
    }
#pragma unroll
    for (int r = 0; r < 8; ++r) {
      int m = r + 8 * hi;
      float ps = 0.0f;
#pragma unroll
      for (int nt = 0; nt < 4; ++nt) {
        float p = __expf(S[nt][r] * 0.125f - nm[r]);   // 1/sqrt(64)
        Psm[wave][m * LCH + nt * 16 + ln] = (bf16)p;   // C-layout -> row-major P
        ps += p;
      }
      ps += __shfl_xor(ps, 1, 32);
      ps += __shfl_xor(ps, 2, 32);
      ps += __shfl_xor(ps, 4, 32);
      ps += __shfl_xor(ps, 8, 32);
      rsum[r] = rsum[r] * corr[r] + ps;
#pragma unroll
      for (int dt = 0; dt < 4; ++dt) O[dt][r] *= corr[r];
    }
    wait_ds0();   // per-wave LDS in-order: P stores complete before fragment reload

    // O += P . V  (K-dim = 64 -> 2 fragment steps)
#pragma unroll
    for (int kk = 0; kk < 2; ++kk) {
      v16bf pfr = load_frag(&Psm[wave][ln * LCH + kk * 32 + hi * 8],
                            &Psm[wave][ln * LCH + kk * 32 + 16 + hi * 8]);
#pragma unroll
      for (int dt = 0; dt < 4; ++dt) {
        int r = dt * 16 + ln;
        v16bf bfr = load_frag(&Vsm[buf][r * LCH + kk * 32 + hi * 8],
                              &Vsm[buf][r * LCH + kk * 32 + 16 + hi * 8]);
        O[dt] = wmma_bf16(pfr, bfr, O[dt]);
      }
    }
    __syncthreads();
    buf ^= 1;
  }

  // normalize and write merged-head output [b, s, h*64+d] as bf16
#pragma unroll
  for (int r = 0; r < 8; ++r) {
    float inv = 1.0f / rsum[r];
    int ss = s0 + wave * 16 + r + 8 * hi;
#pragma unroll
    for (int dt = 0; dt < 4; ++dt) {
      int col = h * HDIM + dt * 16 + ln;
      Ob[((size_t)b * SEQ + ss) * DIM + col] = (bf16)(O[dt][r] * inv);
    }
  }
}

// ---------------- host-side orchestration ----------------

extern "C" void kernel_launch(void* const* d_in, const int* in_sizes, int n_in,
                              void* d_out, int out_size, void* d_ws, size_t ws_size,
                              hipStream_t stream) {
  (void)in_sizes; (void)n_in; (void)out_size; (void)ws_size;

  const float* X_Q   = (const float*)d_in[0];
  const float* X_KV  = (const float*)d_in[1];
  const float* q_w   = (const float*)d_in[2];
  const float* q_b   = (const float*)d_in[3];
  const float* kv_w  = (const float*)d_in[4];
  const float* kv_b  = (const float*)d_in[5];
  const float* out_w = (const float*)d_in[6];
  const float* out_b = (const float*)d_in[7];
  float* out = (float*)d_out;

  // workspace carve-up (bf16 staging buffers)
  char* p = (char*)d_ws;
  auto take = [&](size_t bytes) { char* q = p; p += (bytes + 255) & ~(size_t)255; return q; };
  bf16* Xq_bf  = (bf16*)take((size_t)MB * DIM * 2);
  bf16* Xkv_bf = (bf16*)take((size_t)MB * DIM * 2);
  bf16* qwt    = (bf16*)take((size_t)DIM * DIM * 2);
  bf16* kvwt   = (bf16*)take((size_t)2 * DIM * DIM * 2);
  bf16* owt    = (bf16*)take((size_t)DIM * DIM * 2);
  bf16* Q_bf   = (bf16*)take((size_t)MB * DIM * 2);
  bf16* K_bf   = (bf16*)take((size_t)MB * DIM * 2);
  bf16* Vt_bf  = (bf16*)take((size_t)MB * DIM * 2);
  bf16* At_bf  = (bf16*)take((size_t)MB * DIM * 2);

  const int NEL = MB * DIM;  // 8M elements
  cvt_bf16_kernel<<<2048, 256, 0, stream>>>(X_Q,  Xq_bf,  NEL);
  cvt_bf16_kernel<<<2048, 256, 0, stream>>>(X_KV, Xkv_bf, NEL);
  cvt_transpose_kernel<<<1024, 256, 0, stream>>>(q_w,   qwt,  DIM, DIM);
  cvt_transpose_kernel<<<2048, 256, 0, stream>>>(kv_w,  kvwt, DIM, 2 * DIM);
  cvt_transpose_kernel<<<1024, 256, 0, stream>>>(out_w, owt,  DIM, DIM);

  dim3 blk(256);
  dim3 gq(DIM / BN, MB / BM);                 // (4, 64)
  gemm_bf16_wmma_kernel<<<gq, blk, 0, stream>>>(Xq_bf, qwt, q_b,
                                                Q_bf, nullptr, nullptr,
                                                MB, DIM, DIM, 0);
  dim3 gkv(2 * DIM / BN, MB / BM);            // (8, 64)
  gemm_bf16_wmma_kernel<<<gkv, blk, 0, stream>>>(Xkv_bf, kvwt, kv_b,
                                                 K_bf, Vt_bf, nullptr,
                                                 MB, 2 * DIM, DIM, 1);
  dim3 ga(SEQ / 128, NH, 4);                  // (16, 16, 4)
  attn_wmma_kernel<<<ga, blk, 0, stream>>>(Q_bf, K_bf, Vt_bf, At_bf);

  gemm_bf16_wmma_kernel<<<gq, blk, 0, stream>>>(At_bf, owt, out_b,
                                                nullptr, nullptr, out,
                                                MB, DIM, DIM, 2);
}
